// MultiHeadLatentAttention_13864154431650
// MI455X (gfx1250) — matrix operand order repaired and checked by reading, not compiled, not run on hardware
//
#include <hip/hip_runtime.h>

// ---------------- problem constants ----------------
#define BB   2
#define LL   2048
#define DD   2048
#define NH   16
#define DH   128      // base head dim (also value dim)
#define DC   512
#define DCQ  1536
#define DHR  64       // rope head dim
#define HK   192      // DH + DHR (q/k per-head dim)
#define MM   (BB*LL)  // 4096 token rows

typedef __attribute__((ext_vector_type(16))) __bf16 v16bf;
typedef __attribute__((ext_vector_type(8)))  __bf16 v8bf;
typedef __attribute__((ext_vector_type(8)))  float  v8f;

// ---------------- helpers ----------------
static __device__ __forceinline__ unsigned short f2b(float f) {
  unsigned int u = __float_as_uint(f);
  unsigned int r = u + 0x7FFFu + ((u >> 16) & 1u);   // round-to-nearest-even
  return (unsigned short)(r >> 16);
}
static __device__ __forceinline__ float b2f(unsigned short s) {
  return __uint_as_float(((unsigned int)s) << 16);
}
static __device__ __forceinline__ v16bf mk16(v8bf lo, v8bf hi) {
  union { v16bf v; v8bf p[2]; } u;
  u.p[0] = lo; u.p[1] = hi;
  return u.v;
}
// two 8-element chunks p[0..7], p[16..23] (B-fragment per-lane layout, table order)
static __device__ __forceinline__ v16bf ld16c(const unsigned short* p) {
  return mk16(*(const v8bf*)p, *(const v8bf*)(p + 16));
}
static __device__ __forceinline__ v8f wmma_bf16(v16bf a, v16bf b, v8f c) {
  return __builtin_amdgcn_wmma_f32_16x16x32_bf16(false, a, false, b, (short)0, c,
                                                 false, false);
}
static __device__ __forceinline__ void store_out(unsigned short* p, float v) { *p = f2b(v); }
static __device__ __forceinline__ void store_out(float* p, float v) { *p = v; }

// ---- async global -> LDS copy (CDNA5 ASYNCcnt path), with safe fallback ----
#if __has_builtin(__builtin_amdgcn_global_load_async_to_lds_b128)
#define USE_ASYNC_LDS 1
#else
#define USE_ASYNC_LDS 0
#endif

typedef int b128v __attribute__((vector_size(4 * sizeof(int))));
typedef __attribute__((address_space(1))) b128v* b128v_gptr;
typedef __attribute__((address_space(3))) b128v* b128v_lptr;

static __device__ __forceinline__ void cp16_g2s(unsigned short* l, const unsigned short* g) {
#if USE_ASYNC_LDS
  __builtin_amdgcn_global_load_async_to_lds_b128(
      (b128v_gptr)const_cast<unsigned short*>(g),
      (b128v_lptr)l, /*offset=*/0, /*cpol=*/0);
#else
  *(uint4*)l = *(const uint4*)g;
#endif
}
static __device__ __forceinline__ void sync_tiles() {
#if USE_ASYNC_LDS
  asm volatile("s_wait_asynccnt 0x0" ::: "memory");
#endif
  __syncthreads();
}

// ---------------- fp32 -> bf16 cast ----------------
__global__ __launch_bounds__(256) void cast_f32_bf16(const float* __restrict__ in,
                                                     unsigned short* __restrict__ out,
                                                     int n) {
  int i = (blockIdx.x * 256 + threadIdx.x) * 4;
#pragma unroll
  for (int j = 0; j < 4; ++j)
    if (i + j < n) out[i + j] = f2b(in[i + j]);
}

// ---------------- generic bf16 WMMA GEMM ----------------
// C[M,N] = A[M,K] * B[K,N]; A,B row-major bf16; OUT_T in {unsigned short(bf16), float}
// BM=128 BN=128 BK=64; 256 threads = 8 waves, wave grid 4(M) x 2(N), each wave 32x64.
template <typename OUT_T>
__global__ __launch_bounds__(256) void gemm_bf16(const unsigned short* __restrict__ A,
                                                 const unsigned short* __restrict__ B,
                                                 OUT_T* __restrict__ C,
                                                 int M, int N, int K) {
  __shared__ unsigned short As[128 * 64];    // [row][k]
  __shared__ unsigned short BsT[128 * 64];   // [n][k]
  const int tile_n = blockIdx.x * 128;
  const int tile_m = blockIdx.y * 128;
  const int tid  = threadIdx.x;
  const int wave = tid >> 5;
  const int lane = tid & 31;
  const int wm = wave >> 1;     // 0..3
  const int wn = wave & 1;      // 0..1
  const int half = lane >> 4;   // 0/1
  const int r = lane & 15;

  v8f acc[2][4] = {};

  for (int k0 = 0; k0 < K; k0 += 64) {
    // A tile 128x64: direct row-major copy -> async to LDS
#pragma unroll
    for (int i = 0; i < 4; ++i) {
      int id = tid + i * 256;             // 0..1023
      int row = id >> 3;
      int cc  = (id & 7) * 8;
      cp16_g2s(&As[row * 64 + cc], &A[(size_t)(tile_m + row) * K + k0 + cc]);
      if (k0 + 64 < K)
        __builtin_prefetch(&A[(size_t)(tile_m + row) * K + k0 + 64 + cc], 0, 1);
    }
    // B tile 64x128, store transposed [n][k] (manual: transpose in flight)
#pragma unroll
    for (int i = 0; i < 4; ++i) {
      int id = tid + i * 256;
      int krow = id >> 4;                 // 0..63
      int nc   = (id & 15) * 8;           // 0..120
      uint4 t = *(const uint4*)&B[(size_t)(k0 + krow) * N + tile_n + nc];
      const unsigned short* tp = (const unsigned short*)&t;
#pragma unroll
      for (int j = 0; j < 8; ++j) BsT[(nc + j) * 64 + krow] = tp[j];
      if (k0 + 64 < K)
        __builtin_prefetch(&B[(size_t)(k0 + 64 + krow) * N + tile_n + nc], 0, 1);
    }
    sync_tiles();

#pragma unroll
    for (int kk = 0; kk < 64; kk += 32) {
      v16bf afrag[2], bfrag[4];
#pragma unroll
      for (int mt = 0; mt < 2; ++mt) {
        int row = wm * 32 + mt * 16 + r;
        v8bf lo = *(const v8bf*)&As[row * 64 + kk + half * 8];
        v8bf hi = *(const v8bf*)&As[row * 64 + kk + 16 + half * 8];
        afrag[mt] = mk16(lo, hi);
      }
#pragma unroll
      for (int nt = 0; nt < 4; ++nt) {
        int col = wn * 64 + nt * 16 + r;
        bfrag[nt] = ld16c(&BsT[col * 64 + kk + half * 8]);
      }
#pragma unroll
      for (int mt = 0; mt < 2; ++mt)
#pragma unroll
        for (int nt = 0; nt < 4; ++nt)
          acc[mt][nt] = wmma_bf16(afrag[mt], bfrag[nt], acc[mt][nt]);
    }
    __syncthreads();
  }

  // epilogue: C/D layout row = half*8 + j, col = r
#pragma unroll
  for (int mt = 0; mt < 2; ++mt)
#pragma unroll
    for (int nt = 0; nt < 4; ++nt)
#pragma unroll
      for (int j = 0; j < 8; ++j) {
        int row = tile_m + wm * 32 + mt * 16 + half * 8 + j;
        int col = tile_n + wn * 64 + nt * 16 + r;
        store_out(&C[(size_t)row * N + col], acc[mt][nt][j]);
      }
}

// ---------------- q/k/v assembly + RoPE ----------------
// tqb [M][NH*DH], tqr [M][NH*DHR], tkb [M][NH*DH], tkr [M][NH*DHR], tv [M][NH*DH]
// q,k -> [b][h][l][HK] bf16 (q pre-scaled by 1/sqrt(DH)); v -> [b][h][l][DH]
__global__ __launch_bounds__(256) void assemble_qkv(const unsigned short* __restrict__ tqb,
                                                    const unsigned short* __restrict__ tqr,
                                                    const unsigned short* __restrict__ tkb,
                                                    const unsigned short* __restrict__ tkr,
                                                    const unsigned short* __restrict__ tv,
                                                    unsigned short* __restrict__ q,
                                                    unsigned short* __restrict__ k,
                                                    unsigned short* __restrict__ v) {
  const int m = blockIdx.x;          // b*L + l
  const int b = m >> 11;
  const int l = m & 2047;
  const float scale = 0.08838834764831845f;  // DH^-0.5

  for (int idx = threadIdx.x; idx < NH * DH; idx += 256) {
    int h = idx >> 7, d = idx & 127;
    size_t hb = ((size_t)(b * NH + h) * LL + l);
    q[hb * HK + d] = f2b(b2f(tqb[(size_t)m * (NH * DH) + idx]) * scale);
    k[hb * HK + d] = tkb[(size_t)m * (NH * DH) + idx];
    v[hb * DH + d] = tv[(size_t)m * (NH * DH) + idx];
  }
  // RoPE on the DHR=64 tail: pairs (j, j+32), j in [0,32)
  for (int idx = threadIdx.x; idx < NH * 32; idx += 256) {
    int h = idx >> 5, j = idx & 31;
    float inv = __expf(-(float)j * 0.28782313662425572f);  // 10000^(-j/32)
    float ang = (float)l * inv;
    float cs = cosf(ang), sn = sinf(ang);
    size_t src = (size_t)m * (NH * DHR) + h * DHR + j;
    size_t dst = ((size_t)(b * NH + h) * LL + l) * HK + DH;
    float q0 = b2f(tqr[src]), q1 = b2f(tqr[src + 32]);
    q[dst + j]      = f2b((q0 * cs - q1 * sn) * scale);
    q[dst + j + 32] = f2b((q1 * cs + q0 * sn) * scale);
    float k0 = b2f(tkr[src]), k1 = b2f(tkr[src + 32]);
    k[dst + j]      = f2b(k0 * cs - k1 * sn);
    k[dst + j + 32] = f2b(k1 * cs + k0 * sn);
  }
}

// ---------------- flash attention (non-causal) ----------------
// grid (L/128, NH, B); 256 threads = 8 waves, each wave owns 16 query rows.
#define BKEY 64
__global__ __launch_bounds__(256) void mla_attn(const unsigned short* __restrict__ q,
                                                const unsigned short* __restrict__ k,
                                                const unsigned short* __restrict__ v,
                                                unsigned short* __restrict__ out) {
  __shared__ unsigned short Kls[BKEY * HK];      // [key][dim]   24.0 KB
  __shared__ unsigned short VTls[DH * BKEY];     // [dim][key]   16.0 KB
  __shared__ unsigned short Pst[8 * 16 * BKEY];  // per-wave P   16.0 KB

  const int h = blockIdx.y, b = blockIdx.z;
  const int tid = threadIdx.x, wave = tid >> 5, lane = tid & 31;
  const int half = lane >> 4, r = lane & 15;
  const size_t head = (size_t)(b * NH + h) * LL;
  const int qrow0 = blockIdx.x * 128 + wave * 16;

  // Q fragments: 16 rows x 192 dims = 6 k-steps, kept in VGPRs
  v16bf qfrag[6];
  {
    const unsigned short* qp = q + (head + qrow0 + r) * HK;
#pragma unroll
    for (int kk = 0; kk < 6; ++kk) {
      v8bf lo = *(const v8bf*)&qp[kk * 32 + half * 8];
      v8bf hi = *(const v8bf*)&qp[kk * 32 + 16 + half * 8];
      qfrag[kk] = mk16(lo, hi);
    }
  }

  v8f o[8] = {};                 // 16 x 128 output accumulators
  float mrow[8], lrow[8];
#pragma unroll
  for (int j = 0; j < 8; ++j) { mrow[j] = -3.0e38f; lrow[j] = 0.0f; }

  for (int kb = 0; kb < LL; kb += BKEY) {
    // K tile: 64 keys x 192 dims, direct row-major copy -> async to LDS
#pragma unroll
    for (int i = 0; i < 6; ++i) {
      int id = tid + i * 256;            // 0..1535
      int row = id / 24;
      int cc  = (id % 24) * 8;
      cp16_g2s(&Kls[row * HK + cc], &k[(head + kb + row) * HK + cc]);
      if (kb + BKEY < LL)
        __builtin_prefetch(&k[(head + kb + BKEY + row) * HK + cc], 0, 1);
    }
    // V tile transposed: VTls[dim][key] (manual transpose)
#pragma unroll
    for (int i = 0; i < 4; ++i) {
      int id = tid + i * 256;            // 0..1023
      int krow = id >> 4;
      int nc   = (id & 15) * 8;
      uint4 t = *(const uint4*)&v[(head + kb + krow) * DH + nc];
      const unsigned short* tp = (const unsigned short*)&t;
#pragma unroll
      for (int j = 0; j < 8; ++j) VTls[(nc + j) * BKEY + krow] = tp[j];
      if (kb + BKEY < LL)
        __builtin_prefetch(&v[(head + kb + BKEY + krow) * DH + nc], 0, 1);
    }
    sync_tiles();

    // S = Q K^T  (16 x 64), 4 key tiles x 6 k-steps
    v8f s[4] = {};
#pragma unroll
    for (int nt = 0; nt < 4; ++nt)
#pragma unroll
      for (int kk = 0; kk < 6; ++kk) {
        v16bf bf = ld16c(&Kls[(nt * 16 + r) * HK + kk * 32 + half * 8]);
        s[nt] = wmma_bf16(qfrag[kk], bf, s[nt]);
      }

    // online softmax: rows (half*8+j) live in a 16-lane group
#pragma unroll
    for (int j = 0; j < 8; ++j) {
      float mx = fmaxf(fmaxf(s[0][j], s[1][j]), fmaxf(s[2][j], s[3][j]));
      mx = fmaxf(mx, __shfl_xor(mx, 1));
      mx = fmaxf(mx, __shfl_xor(mx, 2));
      mx = fmaxf(mx, __shfl_xor(mx, 4));
      mx = fmaxf(mx, __shfl_xor(mx, 8));
      float mnew  = fmaxf(mrow[j], mx);
      float alpha = __expf(mrow[j] - mnew);
      float sum = 0.0f;
#pragma unroll
      for (int nt = 0; nt < 4; ++nt) {
        float p = __expf(s[nt][j] - mnew);
        s[nt][j] = p;
        sum += p;
      }
      sum += __shfl_xor(sum, 1);
      sum += __shfl_xor(sum, 2);
      sum += __shfl_xor(sum, 4);
      sum += __shfl_xor(sum, 8);
      lrow[j] = lrow[j] * alpha + sum;
      mrow[j] = mnew;
#pragma unroll
      for (int nt = 0; nt < 8; ++nt) o[nt][j] *= alpha;
    }

    // stage P (16 x 64) -> LDS (C layout -> A layout crossing), per-wave region
    const int wbase = wave * 16 * BKEY;
#pragma unroll
    for (int nt = 0; nt < 4; ++nt)
#pragma unroll
      for (int j = 0; j < 8; ++j)
        Pst[wbase + (half * 8 + j) * BKEY + nt * 16 + r] = f2b(s[nt][j]);

    // O += P V : 2 k-steps over 64 keys, 8 dim tiles
#pragma unroll
    for (int kk2 = 0; kk2 < BKEY; kk2 += 32) {
      v8bf plo = *(const v8bf*)&Pst[wbase + r * BKEY + kk2 + half * 8];
      v8bf phi = *(const v8bf*)&Pst[wbase + r * BKEY + kk2 + 16 + half * 8];
      v16bf pf = mk16(plo, phi);
#pragma unroll
      for (int nt = 0; nt < 8; ++nt) {
        v16bf vf = ld16c(&VTls[(nt * 16 + r) * BKEY + kk2 + half * 8]);
        o[nt] = wmma_bf16(pf, vf, o[nt]);
      }
    }
    __syncthreads();
  }

  // normalize and write attn_out [b*L + row][h*128 + col] (bf16, row-major)
#pragma unroll
  for (int nt = 0; nt < 8; ++nt)
#pragma unroll
    for (int j = 0; j < 8; ++j) {
      int row = qrow0 + half * 8 + j;
      float val = o[nt][j] / lrow[j];
      out[((size_t)b * LL + row) * (NH * DH) + h * DH + nt * 16 + r] = f2b(val);
    }
}

// ---------------- host launcher ----------------
extern "C" void kernel_launch(void* const* d_in, const int* in_sizes, int n_in,
                              void* d_out, int out_size, void* d_ws, size_t ws_size,
                              hipStream_t stream) {
  (void)in_sizes; (void)n_in; (void)out_size; (void)ws_size;
  const float* x_f    = (const float*)d_in[0];
  const float* wdq_f  = (const float*)d_in[1];
  const float* wuq_f  = (const float*)d_in[2];
  const float* wqr_f  = (const float*)d_in[3];
  const float* wdkv_f = (const float*)d_in[4];
  const float* wuk_f  = (const float*)d_in[5];
  const float* wkr_f  = (const float*)d_in[6];
  const float* wuv_f  = (const float*)d_in[7];
  const float* wo_f   = (const float*)d_in[8];
  float* out = (float*)d_out;

  char* ws = (char*)d_ws;
  size_t off = 0;
  auto alloc = [&](size_t elems) {
    unsigned short* p = (unsigned short*)(ws + off);
    off += (elems * 2 + 255) & ~(size_t)255;
    return p;
  };

  unsigned short* xb   = alloc((size_t)MM * DD);
  unsigned short* wdq  = alloc((size_t)DD * DCQ);
  unsigned short* wuq  = alloc((size_t)DCQ * NH * DH);
  unsigned short* wqr  = alloc((size_t)DCQ * NH * DHR);
  unsigned short* wdkv = alloc((size_t)DD * DC);
  unsigned short* wuk  = alloc((size_t)DC * NH * DH);
  unsigned short* wkr  = alloc((size_t)DD * NH * DHR);
  unsigned short* wuv  = alloc((size_t)DC * NH * DH);
  unsigned short* wo   = alloc((size_t)NH * DH * DD);
  unsigned short* qc   = alloc((size_t)MM * DCQ);
  unsigned short* cb   = alloc((size_t)MM * DC);
  unsigned short* tqb  = alloc((size_t)MM * NH * DH);
  unsigned short* tqr  = alloc((size_t)MM * NH * DHR);
  unsigned short* tkb  = alloc((size_t)MM * NH * DH);
  unsigned short* tkr  = alloc((size_t)MM * NH * DHR);
  unsigned short* tv   = alloc((size_t)MM * NH * DH);
  unsigned short* qg   = alloc((size_t)BB * NH * LL * HK);
  unsigned short* kg   = alloc((size_t)BB * NH * LL * HK);
  unsigned short* vg   = alloc((size_t)BB * NH * LL * DH);
  unsigned short* ao   = alloc((size_t)MM * NH * DH);

  // 1) casts
  {
    const float* srcs[9]  = { x_f, wdq_f, wuq_f, wqr_f, wdkv_f, wuk_f, wkr_f, wuv_f, wo_f };
    unsigned short* dsts[9] = { xb, wdq, wuq, wqr, wdkv, wuk, wkr, wuv, wo };
    const int cnts[9] = { MM * DD, DD * DCQ, DCQ * NH * DH, DCQ * NH * DHR,
                          DD * DC, DC * NH * DH, DD * NH * DHR, DC * NH * DH,
                          NH * DH * DD };
    for (int i = 0; i < 9; ++i)
      cast_f32_bf16<<<dim3((cnts[i] + 1023) / 1024), dim3(256), 0, stream>>>(
          srcs[i], dsts[i], cnts[i]);
  }

  dim3 blk(256);
  // 2) projections
  gemm_bf16<unsigned short><<<dim3(DCQ / 128, MM / 128), blk, 0, stream>>>(
      xb, wdq, qc, MM, DCQ, DD);                                  // qc = x @ W_D_Q
  gemm_bf16<unsigned short><<<dim3(DC / 128, MM / 128), blk, 0, stream>>>(
      xb, wdkv, cb, MM, DC, DD);                                  // c = x @ W_D_KV
  gemm_bf16<unsigned short><<<dim3((NH * DH) / 128, MM / 128), blk, 0, stream>>>(
      qc, wuq, tqb, MM, NH * DH, DCQ);                            // q_base
  gemm_bf16<unsigned short><<<dim3((NH * DHR) / 128, MM / 128), blk, 0, stream>>>(
      qc, wqr, tqr, MM, NH * DHR, DCQ);                           // q_rope
  gemm_bf16<unsigned short><<<dim3((NH * DH) / 128, MM / 128), blk, 0, stream>>>(
      cb, wuk, tkb, MM, NH * DH, DC);                             // k_base
  gemm_bf16<unsigned short><<<dim3((NH * DHR) / 128, MM / 128), blk, 0, stream>>>(
      xb, wkr, tkr, MM, NH * DHR, DD);                            // k_rope
  gemm_bf16<unsigned short><<<dim3((NH * DH) / 128, MM / 128), blk, 0, stream>>>(
      cb, wuv, tv, MM, NH * DH, DC);                              // v

  // 3) assemble q/k/v (+RoPE, +scale folded into q)
  assemble_qkv<<<dim3(MM), blk, 0, stream>>>(tqb, tqr, tkb, tkr, tv, qg, kg, vg);

  // 4) flash attention
  mla_attn<<<dim3(LL / 128, NH, BB), blk, 0, stream>>>(qg, kg, vg, ao);

  // 5) output projection (f32 epilogue into d_out)
  gemm_bf16<float><<<dim3(DD / 128, MM / 128), blk, 0, stream>>>(
      ao, wo, out, MM, DD, NH * DH);
}
